// _StratifiedRegTsneModel_39359080301166
// MI455X (gfx1250) — matrix operand order repaired and checked by reading, compiled, not run on hardware
//
#include <hip/hip_runtime.h>
#include <hip/hip_bf16.h>
#include <cstdint>

// ---------------------------------------------------------------------------
// t-SNE KL loss, fused:  loss = 0.5 * sum_b [ A_b + B_b * log(S_b) ]
//   S = sum_{i!=j} 1/(1+d2_ij)
//   A = sum_{i!=j} p_ij * (log p_ij + log(1+d2_ij))
//   B = sum_{i!=j} p_ij
// d2_ij = max(sq_i + sq_j - 2*(Y Y^T)_ij, 0),  Y = X*W
// GEMM done with v_wmma_f32_16x16x32_f16 (f16 operands, f32 accumulate).
// P tile is prefetched (global_prefetch_b8) before the K-loop so the HBM
// stream of P overlaps the WMMA compute phase.
// ---------------------------------------------------------------------------

#define NROWS 4096
#define NFEAT 256

typedef _Float16 half16 __attribute__((ext_vector_type(16)));
typedef float    float8 __attribute__((ext_vector_type(8)));

union FragCast {
    half16 h;
    uint4  u[2];
};

// A-operand (16x32 f16) layout:
//  lanes 0-15 : halves 0-7 -> K = k0+0..7   ; halves 8-15 -> K = k0+16..23
//  lanes16-31 : halves 0-7 -> K = k0+8..15  ; halves 8-15 -> K = k0+24..31
__device__ inline half16 load_frag_A(const _Float16* __restrict__ Y,
                                     int row, int k0, int lane) {
    const int hi = lane >> 4;                       // 0 or 1
    const uint4* p = reinterpret_cast<const uint4*>(Y + (size_t)row * NFEAT + k0 + hi * 8);
    FragCast f;
    f.u[0] = p[0];   // 8 halves at K = k0 + hi*8
    f.u[1] = p[2];   // 8 halves at K = k0 + hi*8 + 16  (+16 halves = 2 uint4)
    return f.h;
}

// B-operand (32x16 f16) layout (column n held by lane n%16):
//  lanes 0-15 : halves 0-7 -> K = k0+0..7   ; halves 8-15 -> K = k0+16..23
//  lanes16-31 : halves 0-7 -> K = k0+8..15  ; halves 8-15 -> K = k0+24..31
__device__ inline half16 load_frag_B(const _Float16* __restrict__ Y,
                                     int col, int k0, int lane) {
    const int hi = lane >> 4;
    const uint4* p = reinterpret_cast<const uint4*>(Y + (size_t)col * NFEAT + k0 + hi * 8);
    FragCast f;
    f.u[0] = p[0];
    f.u[1] = p[2];
    return f.h;
}

// ------------------------- init: zero 6 accumulators ------------------------
__global__ void tsne_init(float* __restrict__ acc) {
    if (threadIdx.x < 6) acc[threadIdx.x] = 0.0f;
}

// ------------- prep: Y = X*W, sq[i] = sum Y^2, Yh = f16(Y) -----------------
__global__ __launch_bounds__(256) void tsne_prep(const float* __restrict__ X1,
                                                 const float* __restrict__ X2,
                                                 const float* __restrict__ W,
                                                 _Float16* __restrict__ Yh,
                                                 float* __restrict__ sq) {
    const int row   = blockIdx.x;
    const int batch = blockIdx.y;
    const int f     = threadIdx.x;

    const float* __restrict__ X = batch ? X2 : X1;
    const float y = X[(size_t)row * NFEAT + f] * W[f];

    Yh[(size_t)batch * NROWS * NFEAT + (size_t)row * NFEAT + f] = (_Float16)y;

    __shared__ float red[256];
    red[f] = y * y;
    __syncthreads();
    for (int s = 128; s > 0; s >>= 1) {
        if (f < s) red[f] += red[f + s];
        __syncthreads();
    }
    if (f == 0) sq[batch * NROWS + row] = red[0];
}

// ------------------- main fused GEMM tile + epilogue -----------------------
// Grid: (32, 32, 2), block 256 threads (8 waves). Tile = 128x128.
// Wave (wM = wave/2 in 0..3, wN = wave%2 in 0..1) covers 32 rows x 64 cols:
//   2 (M) x 4 (N) fragments of 16x16, f32 accumulators.
__global__ __launch_bounds__(256) void tsne_tile(const float* __restrict__ P1,
                                                 const float* __restrict__ P2,
                                                 const _Float16* __restrict__ YhBase,
                                                 const float* __restrict__ sqBase,
                                                 float* __restrict__ acc) {
    const int tid  = threadIdx.x;
    const int lane = tid & 31;
    const int wave = tid >> 5;
    const int wM   = wave >> 1;       // 0..3
    const int wN   = wave & 1;        // 0..1

    const int batch = blockIdx.z;
    const float* __restrict__ P  = batch ? P2 : P1;
    const _Float16* __restrict__ Y = YhBase + (size_t)batch * NROWS * NFEAT;
    const float* __restrict__ sq = sqBase + batch * NROWS;
    float* __restrict__ accb     = acc + batch * 3;

    const int tileM = blockIdx.y * 128;
    const int tileN = blockIdx.x * 128;
    const int rowBase = tileM + wM * 32;
    const int colBase = tileN + wN * 64;
    const int lrow = lane & 15;

    // Prefetch this block's 128x128 P tile (64 KB) so the HBM stream of P
    // overlaps the WMMA phase; the epilogue then hits in cache.
    // 128 rows x 512 B = 512 x 128 B cachelines -> 2 prefetches per thread.
    {
        const float* Ptile = P + (size_t)tileM * NROWS + tileN;
        #pragma unroll
        for (int i = 0; i < 2; ++i) {
            const int idx = tid + i * 256;       // 0..511
            const int r   = idx >> 2;            // row 0..127
            const int cb  = (idx & 3) * 32;      // 32 floats = 128 B
            __builtin_prefetch(Ptile + (size_t)r * NROWS + cb, 0, 1);
        }
    }

    float8 c[2][4];
    const float8 zero8 = {0.f, 0.f, 0.f, 0.f, 0.f, 0.f, 0.f, 0.f};
    for (int i = 0; i < 2; ++i)
        for (int j = 0; j < 4; ++j)
            c[i][j] = zero8;

    for (int k0 = 0; k0 < NFEAT; k0 += 32) {
        half16 a0 = load_frag_A(Y, rowBase + lrow,      k0, lane);
        half16 a1 = load_frag_A(Y, rowBase + 16 + lrow, k0, lane);
        half16 b0 = load_frag_B(Y, colBase +  0 + lrow, k0, lane);
        half16 b1 = load_frag_B(Y, colBase + 16 + lrow, k0, lane);
        half16 b2 = load_frag_B(Y, colBase + 32 + lrow, k0, lane);
        half16 b3 = load_frag_B(Y, colBase + 48 + lrow, k0, lane);

        c[0][0] = __builtin_amdgcn_wmma_f32_16x16x32_f16(false, a0, false, b0,
                                                         (short)0, c[0][0], false, false);
        c[0][1] = __builtin_amdgcn_wmma_f32_16x16x32_f16(false, a0, false, b1,
                                                         (short)0, c[0][1], false, false);
        c[0][2] = __builtin_amdgcn_wmma_f32_16x16x32_f16(false, a0, false, b2,
                                                         (short)0, c[0][2], false, false);
        c[0][3] = __builtin_amdgcn_wmma_f32_16x16x32_f16(false, a0, false, b3,
                                                         (short)0, c[0][3], false, false);
        c[1][0] = __builtin_amdgcn_wmma_f32_16x16x32_f16(false, a1, false, b0,
                                                         (short)0, c[1][0], false, false);
        c[1][1] = __builtin_amdgcn_wmma_f32_16x16x32_f16(false, a1, false, b1,
                                                         (short)0, c[1][1], false, false);
        c[1][2] = __builtin_amdgcn_wmma_f32_16x16x32_f16(false, a1, false, b2,
                                                         (short)0, c[1][2], false, false);
        c[1][3] = __builtin_amdgcn_wmma_f32_16x16x32_f16(false, a1, false, b3,
                                                         (short)0, c[1][3], false, false);
    }

    // Epilogue: C/D layout -> element e of lane L is (M = e + 8*(L>=16), N = L&15)
    const int mOff = (lane >> 4) * 8;
    float sLoc = 0.0f, aLoc = 0.0f, bLoc = 0.0f;

    for (int mf = 0; mf < 2; ++mf) {
        for (int nf = 0; nf < 4; ++nf) {
            const int nIdx = colBase + nf * 16 + lrow;
            const float sqn = sq[nIdx];
            for (int e = 0; e < 8; ++e) {
                const int mIdx = rowBase + mf * 16 + mOff + e;
                if (mIdx != nIdx) {
                    const float g  = c[mf][nf][e];
                    const float d2 = fmaxf(sq[mIdx] + sqn - 2.0f * g, 0.0f);
                    const float t  = 1.0f / (1.0f + d2);
                    sLoc += t;
                    const float p = P[(size_t)mIdx * NROWS + nIdx];
                    aLoc += p * (__logf(p) + __logf(1.0f + d2));
                    bLoc += p;
                }
            }
        }
    }

    // Block reduction (LDS) then one atomic per quantity per block.
    __shared__ float rs[256], ra[256], rb[256];
    rs[tid] = sLoc; ra[tid] = aLoc; rb[tid] = bLoc;
    __syncthreads();
    for (int s = 128; s > 0; s >>= 1) {
        if (tid < s) {
            rs[tid] += rs[tid + s];
            ra[tid] += ra[tid + s];
            rb[tid] += rb[tid + s];
        }
        __syncthreads();
    }
    if (tid == 0) {
        atomicAdd(accb + 0, rs[0]);
        atomicAdd(accb + 1, ra[0]);
        atomicAdd(accb + 2, rb[0]);
    }
}

// ------------------------------- finalize ----------------------------------
__global__ void tsne_final(const float* __restrict__ acc, float* __restrict__ out) {
    if (threadIdx.x == 0) {
        float loss = 0.0f;
        for (int b = 0; b < 2; ++b) {
            const float S = acc[b * 3 + 0];
            const float A = acc[b * 3 + 1];
            const float B = acc[b * 3 + 2];
            loss += A + B * __logf(S);
        }
        out[0] = 0.5f * loss;
    }
}

extern "C" void kernel_launch(void* const* d_in, const int* in_sizes, int n_in,
                              void* d_out, int out_size, void* d_ws, size_t ws_size,
                              hipStream_t stream) {
    (void)in_sizes; (void)n_in; (void)out_size; (void)ws_size;

    const float* P1 = (const float*)d_in[0];
    const float* P2 = (const float*)d_in[1];
    const float* X1 = (const float*)d_in[2];
    const float* X2 = (const float*)d_in[3];
    const float* W  = (const float*)d_in[4];
    float* out = (float*)d_out;

    // Workspace layout:
    //   [0, 4MB)          : Yh, f16, 2 batches x 4096 x 256
    //   [4MB, 4MB+32KB)   : sq, f32, 2 batches x 4096
    //   [4MB+32KB, ...+24): 6 float accumulators (S, A, B per batch)
    char* ws = (char*)d_ws;
    _Float16* Yh = (_Float16*)ws;
    float* sq    = (float*)(ws + (size_t)4 * 1024 * 1024);
    float* acc   = (float*)(ws + (size_t)4 * 1024 * 1024 + 32 * 1024);

    tsne_init<<<1, 32, 0, stream>>>(acc);
    tsne_prep<<<dim3(NROWS, 2), 256, 0, stream>>>(X1, X2, W, Yh, sq);
    tsne_tile<<<dim3(NROWS / 128, NROWS / 128, 2), 256, 0, stream>>>(P1, P2, Yh, sq, acc);
    tsne_final<<<1, 32, 0, stream>>>(acc, out);
}
